// LSTM_emg_base_base_51453708206507
// MI455X (gfx1250) — matrix operand order repaired and checked by reading, not compiled, not run on hardware
//
#include <hip/hip_runtime.h>

#define INPUT_DIM 16
#define H1 5
#define H2 50
#define NCLS 20
#define TLEN 512
#define ROWS 16      // batch rows per workgroup
#define NT 13        // N tiles (208 / 16)
#define GPAD 208     // padded gate dimension (4*H2 -> 208)
#define KPAD 64      // padded K: [h2(50) | h1(5) | 1.0 | 0-pad] -> 64
#define KAUG (H2 + H1)        // 55: h1 slot base is 50, bias slot is 55

typedef __attribute__((ext_vector_type(16))) __bf16 v16bf;
typedef __attribute__((ext_vector_type(8)))  __bf16 v8bf;
typedef __attribute__((ext_vector_type(8)))  float  v8f;

// Fast activations: v_exp_f32 + v_rcp_f32 (no IEEE div expansion).
__device__ __forceinline__ float fastrcp(float x)   { return __builtin_amdgcn_rcpf(x); }
__device__ __forceinline__ float sigf(float x)      { return fastrcp(1.0f + __expf(-x)); }
__device__ __forceinline__ float tanhfast(float x)  { return 1.0f - 2.0f * fastrcp(__expf(2.0f * x) + 1.0f); }

__global__ __launch_bounds__(128)
void lstm2_wmma_kernel(const float* __restrict__ x,
                       const float* __restrict__ w_ih1, const float* __restrict__ w_hh1,
                       const float* __restrict__ b_ih1, const float* __restrict__ b_hh1,
                       const float* __restrict__ w_ih2, const float* __restrict__ w_hh2,
                       const float* __restrict__ b_ih2, const float* __restrict__ b_hh2,
                       const float* __restrict__ fc_w,  const float* __restrict__ fc_b,
                       float* __restrict__ out, float* __restrict__ feat)
{
    __shared__ float s_wih1[4 * H1 * INPUT_DIM];   // 320
    __shared__ float s_whh1[4 * H1 * H1];          // 100
    __shared__ float s_b1[4 * H1];                 // 20 (b_ih1+b_hh1)
    __shared__ float s_fcw[NCLS * H2];             // 1000
    __shared__ float s_fcb[NCLS];                  // 20
    __shared__ float s_h1[2][ROWS * H1];           // ping-pong layer-1 hidden (f32)
    __shared__ float s_g2[ROWS * GPAD];            // layer-2 gate pre-activations
    __shared__ __attribute__((aligned(16))) __bf16 s_h2bf[ROWS * KPAD]; // augmented A: [h2|h1|1|0]
    __shared__ float s_h2f[ROWS * H2];             // final h2 f32 for FC

    const int tid = threadIdx.x;
    const int b0  = blockIdx.x * ROWS;

    // ---- cooperative parameter staging ----
    for (int i = tid; i < 4 * H1 * INPUT_DIM; i += 128) s_wih1[i] = w_ih1[i];
    for (int i = tid; i < 4 * H1 * H1;        i += 128) s_whh1[i] = w_hh1[i];
    for (int i = tid; i < 4 * H1;             i += 128) s_b1[i]   = b_ih1[i] + b_hh1[i];
    for (int i = tid; i < NCLS * H2;          i += 128) s_fcw[i]  = fc_w[i];
    for (int i = tid; i < NCLS;               i += 128) s_fcb[i]  = fc_b[i];
    // Augmented A buffer: zeros everywhere, 1.0 in the bias column (never rewritten).
    for (int i = tid; i < ROWS * KPAD; i += 128) {
        const int col = i & (KPAD - 1);
        s_h2bf[i] = (col == KAUG) ? (__bf16)1.0f : (__bf16)0.0f;
    }
    for (int i = tid; i < ROWS * H1; i += 128) { s_h1[0][i] = 0.0f; s_h1[1][i] = 0.0f; }

    const int wv   = tid >> 5;   // wave id 0..3
    const int lane = tid & 31;
    const int mrow = lane & 15;  // A row selector, B/C column selector
    const int hh   = lane >> 4;  // half-wave selector

    // ---- pre-pack augmented B into per-lane bf16 fragments (register-resident) ----
    // B is KxN = 64 x 208; rows 0..49 = W_hh2^T, 50..54 = W_ih2^T, 55 = (b_ih2+b_hh2), rest 0.
    // B k order used here (contiguous, same as A): lanes 0-15 hold K=0..15 (VGPR j -> K=2j,2j+1),
    // lanes 16-31 hold K=16..31; N = lane&15.
    v16bf Bf[4][2];
    #pragma unroll
    for (int it = 0; it < 4; ++it) {
        const int nt = wv + it * 4;
        const int n  = nt * 16 + mrow;
        const bool vn = (nt < NT) && (n < 4 * H2);
        const float bsum = vn ? (b_ih2[n] + b_hh2[n]) : 0.0f;
        // kt = 0: k = hh*16 + e in [0..31]  -> pure contiguous W_hh2 row
        {
            v16bf f;
            #pragma unroll
            for (int e = 0; e < 16; ++e) {
                const int k = hh * 16 + e;
                f[e] = (__bf16)(vn ? w_hh2[n * H2 + k] : 0.0f);
            }
            Bf[it][0] = f;
        }
        // kt = 1: k = 32 + hh*16 + e; hh=0 half pure W_hh2 (32..47),
        //         hh=1 half mixes W_hh2(48,49) | W_ih2(50..54) | bias(55) | 0
        {
            v16bf f;
            #pragma unroll
            for (int e = 0; e < 16; ++e) {
                const int k = 32 + hh * 16 + e;
                float v = 0.0f;
                if (vn) {
                    if (k < H2)         v = w_hh2[n * H2 + k];
                    else if (k < KAUG)  v = w_ih2[n * H1 + (k - H2)];
                    else if (k == KAUG) v = bsum;
                }
                f[e] = (__bf16)v;
            }
            Bf[it][1] = f;
        }
    }

    // ---- static state ownership for layer-2: row m2o = tid>>3, cols n = (tid&7)+8r ----
    const int m2o = tid >> 3;
    const int nl2 = tid & 7;
    float c2s[7], h2s[7];
    #pragma unroll
    for (int r = 0; r < 7; ++r) { c2s[r] = 0.0f; h2s[r] = 0.0f; }

    float c1 = 0.0f;                      // layer-1 cell (threads tid<80 own (m1,k1))
    const int m1 = tid / H1;
    const int k1 = tid - m1 * H1;

    // ---- layer-1 step: computes h1(tt); writes f32 ping-pong + bf16 augmented A cols ----
    auto layer1_step = [&](int tt) {
        const int p = tt & 1;
        const float* xr = x + ((size_t)(b0 + m1)) * TLEN * INPUT_DIM + (size_t)tt * INPUT_DIM;
        if (k1 == 0 && tt + 8 < TLEN) __builtin_prefetch(xr + 8 * INPUT_DIM, 0, 0);
        float gi = s_b1[k1], gf = s_b1[H1 + k1], gg = s_b1[2 * H1 + k1], go = s_b1[3 * H1 + k1];
        #pragma unroll
        for (int d = 0; d < INPUT_DIM; ++d) {
            const float xv = xr[d];
            gi += s_wih1[ k1           * INPUT_DIM + d] * xv;
            gf += s_wih1[(H1     + k1) * INPUT_DIM + d] * xv;
            gg += s_wih1[(2 * H1 + k1) * INPUT_DIM + d] * xv;
            go += s_wih1[(3 * H1 + k1) * INPUT_DIM + d] * xv;
        }
        #pragma unroll
        for (int j = 0; j < H1; ++j) {
            const float hv = s_h1[1 - p][m1 * H1 + j];
            gi += s_whh1[ k1           * H1 + j] * hv;
            gf += s_whh1[(H1     + k1) * H1 + j] * hv;
            gg += s_whh1[(2 * H1 + k1) * H1 + j] * hv;
            go += s_whh1[(3 * H1 + k1) * H1 + j] * hv;
        }
        const float iv = sigf(gi), fv = sigf(gf), gv = tanhfast(gg), ov = sigf(go);
        c1 = fv * c1 + iv * gv;
        const float h = ov * tanhfast(c1);
        s_h1[p][m1 * H1 + k1] = h;
        s_h2bf[m1 * KPAD + H2 + k1] = (__bf16)h;   // augmented A columns 50..54
    };

    __syncthreads();
    if (tid < ROWS * H1) layer1_step(0);          // peel t=0
    __syncthreads();

    for (int t = 0; t < TLEN; ++t) {
        // ---- fused pre-activation: g2 = [h2|h1|1] @ [W_hh2^T; W_ih2^T; b2] via WMMA, C = 0 ----
        {
            const v8bf a00 = *(const v8bf*)&s_h2bf[mrow * KPAD +      hh * 16];
            const v8bf a01 = *(const v8bf*)&s_h2bf[mrow * KPAD + 8 + hh * 16];
            const v8bf a10 = *(const v8bf*)&s_h2bf[mrow * KPAD + 32 + hh * 16];
            const v8bf a11 = *(const v8bf*)&s_h2bf[mrow * KPAD + 40 + hh * 16];
            const v16bf A0 = __builtin_shufflevector(a00, a01, 0,1,2,3,4,5,6,7,8,9,10,11,12,13,14,15);
            const v16bf A1 = __builtin_shufflevector(a10, a11, 0,1,2,3,4,5,6,7,8,9,10,11,12,13,14,15);
            #pragma unroll
            for (int it = 0; it < 4; ++it) {
                const int nt = wv + it * 4;
                if (nt < NT) {                 // wave-uniform guard: EXEC all-ones inside
                    const int col = nt * 16 + mrow;
                    v8f c = {};                // SRC2 folds to inline 0
                    c = __builtin_amdgcn_wmma_f32_16x16x32_bf16(false, A0, false, Bf[it][0],
                                                                (short)0, c, false, false);
                    c = __builtin_amdgcn_wmma_f32_16x16x32_bf16(false, A1, false, Bf[it][1],
                                                                (short)0, c, false, false);
                    #pragma unroll
                    for (int v = 0; v < 8; ++v) s_g2[(v + hh * 8) * GPAD + col] = c[v];
                }
            }
        }
        __syncthreads();

        // ---- merged phase: layer-2 gates(t)  ||  layer-1(t+1)  (disjoint LDS regions) ----
        {
            #pragma unroll
            for (int r = 0; r < 7; ++r) {
                const int n = nl2 + (r << 3);
                if (n < H2) {
                    const float gi = s_g2[m2o * GPAD + n];
                    const float gf = s_g2[m2o * GPAD + H2 + n];
                    const float gg = s_g2[m2o * GPAD + 2 * H2 + n];
                    const float go = s_g2[m2o * GPAD + 3 * H2 + n];
                    const float iv = sigf(gi), fv = sigf(gf), gv = tanhfast(gg), ov = sigf(go);
                    const float c = fv * c2s[r] + iv * gv;
                    c2s[r] = c;
                    const float h = ov * tanhfast(c);
                    h2s[r] = h;
                    s_h2bf[m2o * KPAD + n] = (__bf16)h;   // h2 state for next step's A
                }
            }
        }
        if (t + 1 < TLEN && tid < ROWS * H1) layer1_step(t + 1);
        __syncthreads();
    }

    // ---- epilogue: feat = h2[T-1], out = feat @ fc_w^T + fc_b ----
    {
        #pragma unroll
        for (int r = 0; r < 7; ++r) {
            const int n = nl2 + (r << 3);
            if (n < H2) {
                s_h2f[m2o * H2 + n] = h2s[r];
                feat[(size_t)(b0 + m2o) * H2 + n] = h2s[r];
            }
        }
    }
    __syncthreads();
    for (int o = tid; o < ROWS * NCLS; o += 128) {
        const int m = o / NCLS, jc = o - m * NCLS;
        float acc = s_fcb[jc];
        #pragma unroll 10
        for (int d = 0; d < H2; ++d) acc += s_h2f[m * H2 + d] * s_fcw[jc * H2 + d];
        out[(size_t)(b0 + m) * NCLS + jc] = acc;
    }
}

extern "C" void kernel_launch(void* const* d_in, const int* in_sizes, int n_in,
                              void* d_out, int out_size, void* d_ws, size_t ws_size,
                              hipStream_t stream) {
    (void)in_sizes; (void)n_in; (void)out_size; (void)d_ws; (void)ws_size;
    const float* x     = (const float*)d_in[0];
    const float* w_ih1 = (const float*)d_in[1];
    const float* w_hh1 = (const float*)d_in[2];
    const float* b_ih1 = (const float*)d_in[3];
    const float* b_hh1 = (const float*)d_in[4];
    const float* w_ih2 = (const float*)d_in[5];
    const float* w_hh2 = (const float*)d_in[6];
    const float* b_ih2 = (const float*)d_in[7];
    const float* b_hh2 = (const float*)d_in[8];
    const float* fc_w  = (const float*)d_in[9];
    const float* fc_b  = (const float*)d_in[10];

    float* out  = (float*)d_out;                    // (4096, 20)
    float* feat = (float*)d_out + 4096 * NCLS;      // (4096, 50)

    dim3 grid(4096 / ROWS);   // 256 workgroups
    dim3 block(128);          // 4 wave32s
    lstm2_wmma_kernel<<<grid, block, 0, stream>>>(x, w_ih1, w_hh1, b_ih1, b_hh1,
                                                  w_ih2, w_hh2, b_ih2, b_hh2,
                                                  fc_w, fc_b, out, feat);
}
